// S2ConvNet_13726715478138
// MI455X (gfx1250) — matrix operand order repaired and checked by reading, not compiled, not run on hardware
//
#include <hip/hip_runtime.h>
#include <math.h>

#define DEV __device__ __forceinline__

typedef __attribute__((ext_vector_type(16))) _Float16 v16h;
typedef __attribute__((ext_vector_type(8)))  _Float16 v8h;
typedef __attribute__((ext_vector_type(8)))  float    v8f;

constexpr float  PIF = 3.14159265358979323846f;
constexpr double PID = 3.14159265358979323846;

DEV float2 cmac(float2 acc, float2 a, float2 b){
  acc.x += a.x*b.x - a.y*b.y;
  acc.y += a.x*b.y + a.y*b.x;
  return acc;
}

__host__ __device__ constexpr int osq(int l){ return l*(2*l-1)*(2*l+1)/3; }   // sum_{l'<l} (2l'+1)^2
DEV void decode_sq(int e, int& l, int& mh, int& nh){
  int li=0; while (osq(li+1) <= e) li++;
  int r = e - osq(li); int w = 2*li+1;
  l=li; mh=r/w; nh=r%w;
}

// ------------------- constant precompute (device) -------------------
DEV double wignerd(int l, double beta, int mp, int m){
  double cb = cos(beta*0.5), sb = sin(beta*0.5);
  double pref = 0.5*(lgamma((double)(l+mp+1))+lgamma((double)(l-mp+1))
                   +lgamma((double)(l+m+1)) +lgamma((double)(l-m+1)));
  int s0 = (m-mp>0)?(m-mp):0;
  int s1 = (l+m < l-mp)?(l+m):(l-mp);
  double tot=0.0;
  for (int s=s0; s<=s1; ++s){
    double den = lgamma((double)(l+m-s+1))+lgamma((double)(s+1))
               +lgamma((double)(mp-m+s+1))+lgamma((double)(l-mp-s+1));
    double sign = ((mp-m+s)&1)? -1.0:1.0;
    tot += sign*exp(pref-den)*pow(cb,(double)(2*l+m-mp-2*s))*pow(sb,(double)(mp-m+2*s));
  }
  return tot;
}
DEV double quadw(int b, int j){
  double s=0;
  for (int k=0;k<b;k++) s += sin((2*j+1)*(2*k+1)*PID/(4.0*b))/(double)(2*k+1);
  return 2.0/b*sin(PID*(2*j+1)/(4.0*b))*s;
}

__global__ void k_precompute(float* dst, int table, int n){
  int i = blockIdx.x*blockDim.x+threadIdx.x;
  if (i>=n) return;
  switch(table){
    case 0: dst[i] = (float)quadw(3,i); break;            // w3[6]
    case 1: { // dl1 [l<10][j<26][mp] : d^l(beta_j)[mp,0]*w13[j]
      int l=0; while(26*(l+1)*(l+1) <= i) l++;
      int r = i - 26*l*l; int w=2*l+1; int j=r/w, mp=r%w;
      double beta = (j+0.5)*PID/26.0;
      dst[i] = (float)(wignerd(l,beta,mp-l,0)*quadw(13,j));
    } break;
    case 2: { // basis1 complex [l<10][g<32][mp]
      int l=0; while(32*(l+1)*(l+1) <= i) l++;
      int r = i - 32*l*l; int w=2*l+1; int g=r/w, mp=r%w;
      double d = wignerd(l,PID*0.5,mp-l,0);
      double ang = -2.0*PID*(double)g/32.0*(double)(mp-l);
      ((float2*)dst)[i] = make_float2((float)(cos(ang)*d),(float)(sin(ang)*d));
    } break;
    case 3: case 4: case 6: case 7: case 9: { // dmat tables [l][j][m][n]
      int B2 = (table==3||table==4)?20 : (table==9?6:12);
      int l=0; while(B2*osq(l+1) <= i) l++;
      int r = i - B2*osq(l); int w=2*l+1;
      int j = r/(w*w); int rr = r%(w*w); int mh=rr/w, nh=rr%w;
      double beta = (j+0.5)*PID/(double)B2;
      double scale = (table==4)? quadw(10,j) : (table==7)? quadw(6,j) : (double)(2*l+1)*0.5;
      dst[i] = (float)(wignerd(l,beta,mh-l,nh-l)*scale);
    } break;
    case 5: case 8: { // basis complex [l][g<32][m][n]
      int l=0; while(32*osq(l+1) <= i) l++;
      int r = i - 32*osq(l); int w=2*l+1;
      int g = r/(w*w); int rr=r%(w*w); int mh=rr/w, nh=rr%w;
      double d = wignerd(l,PID*0.5,mh-l,nh-l);
      double ang = -2.0*PID*(double)g/32.0*(double)(mh-l);
      ((float2*)dst)[i] = make_float2((float)(cos(ang)*d),(float)(sin(ang)*d));
    } break;
  }
}

// ------------------- layer 1 front end -------------------
__global__ void k_fft1(const float* __restrict__ x, float2* __restrict__ X1){
  const int TOT = 128*34*26*26;
  int idx = blockIdx.x*blockDim.x + threadIdx.x;
  if (idx >= TOT) return;
  int mm = idx % 26; int row = idx / 26;
  const float* xr = x + (size_t)row*26;
  float cr=0.f, ci=0.f;
  float w = -2.0f*PIF*(float)mm/26.0f;
  for (int a=0;a<26;a++){ float ang=w*(float)a; cr += xr[a]*__cosf(ang); ci += xr[a]*__sinf(ang); }
  X1[idx] = make_float2(cr,ci);
}

// fh1[b,f,lin]  lin = l*l + mp  (l<10)
__global__ void k_fh1(const float2* __restrict__ X1, const float* __restrict__ dl1,
                      float2* __restrict__ fh1){
  const int TOT = 128*34*100;
  int idx = blockIdx.x*blockDim.x + threadIdx.x;
  if (idx >= TOT) return;
  int lin = idx%100; int bf = idx/100;
  int l=0; while((l+1)*(l+1)<=lin) l++;
  int mp = lin - l*l; int w = 2*l+1;
  int m = mp - l;
  int mm = ((m % 26) + 26) % 26;
  const float2* Xr = X1 + (size_t)bf*26*26;
  const float* dl = dl1 + 26*l*l;
  float2 acc = make_float2(0.f,0.f);
  for (int j=0;j<26;j++){
    float d = dl[j*w+mp];
    float2 v = Xr[j*26+mm];
    acc.x += v.x*d; acc.y += v.y*d;
  }
  fh1[idx]=acc;
}

// khc1[f,o,lin] = SC1 * sum_g k1[f,o,g]*conj(basis1[l][g][mp])
__global__ void k_khc1(const float* __restrict__ k1, const float2* __restrict__ bas1,
                       float2* __restrict__ khc1, float SC1){
  const int TOT = 34*20*100;
  int idx = blockIdx.x*blockDim.x + threadIdx.x;
  if (idx >= TOT) return;
  int lin = idx%100; int fo = idx/100;
  int l=0; while((l+1)*(l+1)<=lin) l++;
  int mp = lin - l*l; int w = 2*l+1;
  const float2* bp = bas1 + 32*l*l + mp;
  const float* kp = k1 + (size_t)fo*32;
  float2 acc = make_float2(0.f,0.f);
  for (int g=0;g<32;g++){
    float kv = kp[g]*SC1; float2 b = bp[g*w];
    acc.x += kv*b.x; acc.y -= kv*b.y;
  }
  khc1[idx]=acc;
}

// ------------------- fused SO(3) synthesis + bias + relu -------------------
template<int B2, int NSQ>
__global__ void k_synth(const float2* __restrict__ zs, const float* __restrict__ dS,
                        const float* __restrict__ bias, float* __restrict__ out, int O){
  __shared__ float2 s_zs[NSQ];
  __shared__ float  s_dl[NSQ];
  __shared__ float2 s_T[B2*B2];
  __shared__ float2 s_U[B2*B2];
  int j  = blockIdx.x % B2;
  int bo = blockIdx.x / B2;
  int o  = bo % O;
  const float2* zp = zs + (size_t)bo*NSQ;
  for (int e=threadIdx.x; e<NSQ; e+=blockDim.x){
    s_zs[e]=zp[e];
    int l,mh,nh; decode_sq(e,l,mh,nh);
    int w=2*l+1;
    s_dl[e] = dS[B2*osq(l) + j*w*w + mh*w + nh];
  }
  for (int p=threadIdx.x; p<B2*B2; p+=blockDim.x) s_T[p]=make_float2(0.f,0.f);
  __syncthreads();
  for (int e=threadIdx.x; e<NSQ; e+=blockDim.x){
    int l,mh,nh; decode_sq(e,l,mh,nh);
    int p = (((mh-l)%B2)+B2)%B2, q=(((nh-l)%B2)+B2)%B2;
    float d = s_dl[e]; float2 z = s_zs[e];
    atomicAdd(&s_T[p*B2+q].x, z.x*d);
    atomicAdd(&s_T[p*B2+q].y, z.y*d);
  }
  __syncthreads();
  for (int t=threadIdx.x; t<B2*B2; t+=blockDim.x){   // IDFT over q
    int p=t/B2, c=t%B2;
    float2 acc=make_float2(0.f,0.f);
    for (int q=0;q<B2;q++){
      float ang = 2.0f*PIF*(float)(q*c)/(float)B2;
      float2 w = make_float2(__cosf(ang), __sinf(ang));
      acc = cmac(acc, s_T[p*B2+q], w);
    }
    s_U[t]=acc;
  }
  __syncthreads();
  float bb = bias[o];
  float* op = out + (size_t)blockIdx.x * (B2*B2);
  for (int t=threadIdx.x; t<B2*B2; t+=blockDim.x){   // IDFT over p, real part
    int a=t/B2, c=t%B2;
    float acc=0.f;
    for (int p=0;p<B2;p++){
      float ang = 2.0f*PIF*(float)(p*a)/(float)B2;
      acc += s_U[p*B2+c].x*__cosf(ang) - s_U[p*B2+c].y*__sinf(ang);
    }
    float v = acc/(float)(B2*B2) + bb;
    op[t] = v>0.f ? v : 0.f;
  }
}

// ------------------- forward fft2 + freq gather -------------------
template<int B2, int NF>
__global__ void k_fft2g(const float* __restrict__ in, float2* __restrict__ out){
  __shared__ float  s_h[B2*B2];
  __shared__ float2 s_V[B2*NF];
  const int H = NF/2;
  const float* ip = in + (size_t)blockIdx.x*B2*B2;
  for (int t=threadIdx.x; t<B2*B2; t+=blockDim.x) s_h[t]=ip[t];
  __syncthreads();
  for (int t=threadIdx.x; t<B2*NF; t+=blockDim.x){
    int a=t/NF, qi=t%NF; int qv = qi-H;
    float ar=0.f, ai=0.f;
    for (int c=0;c<B2;c++){
      float ang = -2.0f*PIF*(float)(qv*c)/(float)B2;
      float h=s_h[a*B2+c];
      ar += h*__cosf(ang); ai += h*__sinf(ang);
    }
    s_V[t]=make_float2(ar,ai);
  }
  __syncthreads();
  float2* op = out + (size_t)blockIdx.x*NF*NF;
  for (int t=threadIdx.x; t<NF*NF; t+=blockDim.x){
    int pi=t/NF, qi=t%NF; int pv=pi-H;
    float2 acc=make_float2(0.f,0.f);
    for (int a=0;a<B2;a++){
      float ang = -2.0f*PIF*(float)(pv*a)/(float)B2;
      float2 w = make_float2(__cosf(ang), __sinf(ang));
      acc = cmac(acc, s_V[a*NF+qi], w);
    }
    op[t]=acc;
  }
}

// ------------------- so3 analysis: fh / khc -------------------
__global__ void k_fh23(const float2* __restrict__ Xg, const float* __restrict__ dA,
                       float2* __restrict__ fh, int B2, int NF, int NSQ, int TOT){
  int idx = blockIdx.x*blockDim.x + threadIdx.x;
  if (idx >= TOT) return;
  int e = idx%NSQ; int bf = idx/NSQ;
  int l,mh,nh; decode_sq(e,l,mh,nh);
  int w=2*l+1; int H=NF/2;
  int mi = mh-l+H, ni = nh-l+H;
  const float2* xp = Xg + (size_t)bf*B2*NF*NF + mi*NF + ni;
  const float* dp = dA + B2*osq(l) + mh*w + nh;
  float2 acc=make_float2(0.f,0.f);
  for (int j=0;j<B2;j++){
    float d = dp[j*w*w];
    float2 v = xp[(size_t)j*NF*NF];
    acc.x += v.x*d; acc.y += v.y*d;
  }
  fh[idx]=acc;
}

__global__ void k_khc23(const float* __restrict__ kmat, const float2* __restrict__ basis,
                        float2* __restrict__ khc, float SC, int NSQ, int TOT){
  int idx = blockIdx.x*blockDim.x + threadIdx.x;
  if (idx >= TOT) return;
  int e = idx%NSQ; int fo = idx/NSQ;
  int l,nh,kh; decode_sq(e,l,nh,kh);
  int w=2*l+1;
  const float2* bp = basis + 32*osq(l) + nh*w + kh;
  const float* kp = kmat + (size_t)fo*32;
  float2 acc=make_float2(0.f,0.f);
  for (int g=0;g<32;g++){
    float kv = kp[g]*SC; float2 b = bp[g*w*w];
    acc.x += kv*b.x; acc.y -= kv*b.y;   // conj
  }
  khc[idx]=acc;
}

// ------------------- complex-GEMM packing for WMMA -------------------
// A' rows: (b*w + mh), cols k' in [0,2*F*wk) : [fx | fy], zero-padded to Kp
__global__ void k_packA(const float2* __restrict__ fh, _Float16* __restrict__ Ah,
                        int F, int NSQf, int baseA, int w, int mstride, int wk,
                        int Kp, int TOT){
  int idx = blockIdx.x*blockDim.x + threadIdx.x;
  if (idx >= TOT) return;
  int kp = idx % Kp; int row = idx / Kp;
  int b = row / w; int mh = row % w;
  float v = 0.f;
  int KF = F*wk;
  if (kp < 2*KF){
    int comp = kp / KF; int kk = kp % KF;
    int f = kk / wk; int kh = kk % wk;
    float2 z = fh[((size_t)b*F + f)*NSQf + baseA + mh*mstride + kh];
    v = comp ? z.y : z.x;
  }
  Ah[idx] = (_Float16)v;
}

// B rows: (o*w + nh) (padded to Nt), cols k' : mode0 [kx | -ky], mode1 [ky | kx]
__global__ void k_packB(const float2* __restrict__ khc, _Float16* __restrict__ Bh,
                        int F, int O, int NSQk, int baseB, int w, int nstride, int wk,
                        int Kp, int N, int imagMode, int TOT){
  int idx = blockIdx.x*blockDim.x + threadIdx.x;
  if (idx >= TOT) return;
  int kp = idx % Kp; int row = idx / Kp;
  float v = 0.f;
  int KF = F*wk;
  if (row < N && kp < 2*KF){
    int o = row / w; int nh = row % w;
    int comp = kp / KF; int kk = kp % KF;
    int f = kk / wk; int kh = kk % wk;
    float2 z = khc[((size_t)f*O + o)*NSQk + baseB + nh*nstride + kh];
    if (imagMode==0) v = comp ? -z.y : z.x;
    else             v = comp ?  z.x : z.y;
  }
  Bh[idx] = (_Float16)v;
}

__global__ void k_combine(const float* __restrict__ Cr, const float* __restrict__ Ci,
                          float2* __restrict__ zs, int O, int NSQz, int baseZ,
                          int w, int N, int TOT){
  int idx = blockIdx.x*blockDim.x + threadIdx.x;
  if (idx >= TOT) return;
  int col = idx % N; int row = idx / N;
  int b = row/w; int mh = row%w; int o = col/w; int nh = col%w;
  zs[((size_t)b*O + o)*NSQz + baseZ + mh*w + nh] = make_float2(Cr[idx], Ci[idx]);
}

// ------------------- WMMA GEMM: Y[M,N] = A[M,K] * B[Nt,K]^T (+bias) -------------------
// block = 4 waves, computes 64(M) x 16(N) tile; grid (Nt/16, M/64).
// B k-slice staged once per block in LDS via async global->LDS copies (4x reuse),
// A staged alongside. M mult of 64, K mult of 32, B padded to Nt=ceil16(N) rows.
__global__ void __launch_bounds__(128)
k_gemm_wmma(const _Float16* __restrict__ A, const _Float16* __restrict__ B,
            const float* __restrict__ bias, float* __restrict__ Y,
            int M, int N, int K, int hasBias){
  __shared__ __align__(64) _Float16 sA[64*32];   // 64 rows x 32 halves (64B/row)
  __shared__ __align__(64) _Float16 sB[16*32];   // 16 rows x 32 halves
  int tid  = threadIdx.x;
  int lane = tid & 31;
  int wave = tid >> 5;
  bool hi  = lane >= 16;
  int r15  = lane & 15;
  int mBase = blockIdx.y*64;
  int nBase = blockIdx.x*16;
  unsigned ldsA = (unsigned)(uintptr_t)(&sA[0]);
  unsigned ldsB = (unsigned)(uintptr_t)(&sB[0]);
  v8f acc = {};
  for (int k0=0; k0<K; k0+=32){
    // ---- stage A (64x32, 256x16B) and B (16x32, 64x16B) via async copies ----
    {
      int r0 = tid >> 2, c0 = tid & 3;                // A chunks 0..127
      const _Float16* g0 = A + (size_t)(mBase + r0)*K + k0 + c0*8;
      unsigned l0 = ldsA + (unsigned)(r0*64 + c0*16);
      asm volatile("global_load_async_to_lds_b128 %0, %1, off" :: "v"(l0), "v"(g0) : "memory");
      int t1 = tid + 128;                              // A chunks 128..255
      int r1 = t1 >> 2, c1 = t1 & 3;
      const _Float16* g1 = A + (size_t)(mBase + r1)*K + k0 + c1*8;
      unsigned l1 = ldsA + (unsigned)(r1*64 + c1*16);
      asm volatile("global_load_async_to_lds_b128 %0, %1, off" :: "v"(l1), "v"(g1) : "memory");
      if (tid < 64){                                   // B chunks 0..63
        int rb = tid >> 2, cb = tid & 3;
        const _Float16* g2 = B + (size_t)(nBase + rb)*K + k0 + cb*8;
        unsigned l2 = ldsB + (unsigned)(rb*64 + cb*16);
        asm volatile("global_load_async_to_lds_b128 %0, %1, off" :: "v"(l2), "v"(g2) : "memory");
      }
      asm volatile("s_wait_asynccnt 0" ::: "memory");
      __syncthreads();
    }
    // ---- fragments from LDS ----
    int arow = wave*16 + r15;
    const _Float16* pa = sA + arow*32 + (hi?16:0);
    v8h a0 = *(const v8h*)(pa);
    v8h a1 = *(const v8h*)(pa + 8);
    v16h bv = *(const v16h*)(sB + r15*32 + (hi?16:0));
    v16h av;
#pragma unroll
    for (int i=0;i<8;i++){ av[i]=a0[i]; av[i+8]=a1[i]; }
    acc = __builtin_amdgcn_wmma_f32_16x16x32_f16(false, av, false, bv, (short)0, acc, false, false);
    __syncthreads();   // protect LDS before next stage overwrites
  }
  int col = nBase + r15;
  if (col < N){
    float bb = hasBias ? bias[col] : 0.f;
#pragma unroll
    for (int r=0;r<8;r++){
      int mr = mBase + wave*16 + r + (hi?8:0);
      Y[(size_t)mr*N + col] = acc[r] + bb;
    }
  }
}

// ------------------- weight packing for MLP -------------------
__global__ void k_packW(const float* __restrict__ W, _Float16* __restrict__ Wt,
                        int K, int N, int Nt){
  int idx = blockIdx.x*blockDim.x + threadIdx.x;
  int TOT = Nt*K;
  if (idx >= TOT) return;
  int k = idx % K; int n = idx / K;
  Wt[idx] = (_Float16)((n < N) ? W[(size_t)k*N + n] : 0.f);
}

// ------------------- integrate (writes f16 activations) -------------------
__global__ void k_integrate(const float* __restrict__ h3, const float* __restrict__ w3,
                            _Float16* __restrict__ feat){
  const int TOT = 128*256;
  int idx = blockIdx.x*blockDim.x + threadIdx.x;
  if (idx >= TOT) return;
  const float* hp = h3 + (size_t)idx*216;
  float acc=0.f;
  for (int j=0;j<6;j++){
    float s=0.f;
    for (int t=0;t<36;t++) s += hp[j*36+t];
    acc += w3[j]*s;
  }
  feat[idx] = (_Float16)(acc * (PIF/3.0f)*(PIF/3.0f));
}

// ------------------- batch-norm -------------------
__global__ void k_bn_relu_h(const float* __restrict__ Yin, const float* __restrict__ g,
                            const float* __restrict__ be, _Float16* __restrict__ out, int N){
  __shared__ float red[128];
  int f = blockIdx.x; int t = threadIdx.x;   // blockDim == 128 == batch
  float v = Yin[(size_t)t*N + f];
  red[t]=v; __syncthreads();
  for (int s=64;s>0;s>>=1){ if (t<s) red[t]+=red[t+s]; __syncthreads(); }
  float mu = red[0]/128.f; __syncthreads();
  float d = v-mu; red[t]=d*d; __syncthreads();
  for (int s=64;s>0;s>>=1){ if (t<s) red[t]+=red[t+s]; __syncthreads(); }
  float var = red[0]/128.f;
  float y = d * rsqrtf(var+1e-5f) * g[f] + be[f];
  out[(size_t)t*N + f] = (_Float16)fmaxf(y,0.f);
}

__global__ void k_bn_sig(const float* __restrict__ Yin, const float* __restrict__ g,
                         const float* __restrict__ be, float* __restrict__ out, int N){
  __shared__ float red[128];
  int f = blockIdx.x; int t = threadIdx.x;
  float v = Yin[(size_t)t*N + f];
  red[t]=v; __syncthreads();
  for (int s=64;s>0;s>>=1){ if (t<s) red[t]+=red[t+s]; __syncthreads(); }
  float mu = red[0]/128.f; __syncthreads();
  float d = v-mu; red[t]=d*d; __syncthreads();
  for (int s=64;s>0;s>>=1){ if (t<s) red[t]+=red[t+s]; __syncthreads(); }
  float var = red[0]/128.f;
  float y = d * rsqrtf(var+1e-5f) * g[f] + be[f];
  out[(size_t)t*N + f] = 1.f/(1.f+__expf(-y));
}

// ------------------- host orchestration -------------------
extern "C" void kernel_launch(void* const* d_in, const int* in_sizes, int n_in,
                              void* d_out, int out_size, void* d_ws, size_t ws_size,
                              hipStream_t stream) {
  (void)in_sizes; (void)n_in; (void)out_size; (void)ws_size;
  const float* x   = (const float*)d_in[0];
  const float* k1  = (const float*)d_in[1];
  const float* b1  = (const float*)d_in[2];
  const float* k2  = (const float*)d_in[3];
  const float* b2  = (const float*)d_in[4];
  const float* k3  = (const float*)d_in[5];
  const float* b3  = (const float*)d_in[6];
  const float* w1  = (const float*)d_in[7];
  const float* bw1 = (const float*)d_in[8];
  const float* w2  = (const float*)d_in[9];
  const float* bw2 = (const float*)d_in[10];
  const float* mw3 = (const float*)d_in[11];
  const float* bw3 = (const float*)d_in[12];
  const float* g1  = (const float*)d_in[13];
  const float* be1 = (const float*)d_in[14];
  const float* g2  = (const float*)d_in[15];
  const float* be2 = (const float*)d_in[16];
  const float* g3  = (const float*)d_in[17];
  const float* be3 = (const float*)d_in[18];

  const float SC1 = (float)(1.0/sqrt(32.0*34.0*1000.0/2197.0));
  const float SC2 = (float)(1.0/sqrt(32.0*20.0*216.0/1000.0));
  const float SC3 = (float)(1.0/sqrt(32.0*128.0*27.0/216.0));

  char* ws = (char*)d_ws;
  size_t off = 0;
  auto alloc = [&](size_t bytes)->size_t{
    size_t o = off; off += (bytes + 255) & ~size_t(255); return o;
  };
  // constants
  float*  c_w3   = (float*) (ws + alloc(6*4));
  float*  c_dl1  = (float*) (ws + alloc(2600*4));
  float2* c_bas1 = (float2*)(ws + alloc(3200*8));
  float*  c_dS10 = (float*) (ws + alloc(26600*4));
  float*  c_dA2  = (float*) (ws + alloc(5720*4));
  float2* c_bas2 = (float2*)(ws + alloc(9152*8));
  float*  c_dS6  = (float*) (ws + alloc(3432*4));
  float*  c_dA3  = (float*) (ws + alloc(420*4));
  float2* c_bas3 = (float2*)(ws + alloc(1120*8));
  float*  c_dS3  = (float*) (ws + alloc(210*4));
  // aliased big regions
  float*  R1 = (float*) (ws + alloc((size_t)28311552*4)); // h1 -> h2 -> h3
  float2* R2 = (float2*)(ws + alloc((size_t)6195200*8));  // X1 -> Xg2 -> Xg3
  float2* R3 = (float2*)(ws + alloc((size_t)4685824*8));  // zs1 -> zs2 -> zs3
  float2* R4a= (float2*)(ws + alloc((size_t)732160*8));   // fh slot
  float2* R4b= (float2*)(ws + alloc((size_t)1146880*8));  // khc slot
  // WMMA scratch
  _Float16* sAh = (_Float16*)(ws + alloc((size_t)819200*2));   // A' pack
  _Float16* sBr = (_Float16*)(ws + alloc((size_t)1638400*2));  // B real pack
  _Float16* sBi = (_Float16*)(ws + alloc((size_t)1638400*2));  // B imag pack
  float*    sCr = (float*)   (ws + alloc((size_t)1982464*4));
  float*    sCi = (float*)   (ws + alloc((size_t)1982464*4));
  // MLP buffers
  _Float16* w1t  = (_Float16*)(ws + alloc(32768*2));
  _Float16* w2t  = (_Float16*)(ws + alloc(4096*2));
  _Float16* w3t  = (_Float16*)(ws + alloc(512*2));
  _Float16* feat = (_Float16*)(ws + alloc(32768*2));
  float*    y1   = (float*)   (ws + alloc(16384*4));
  _Float16* a1h  = (_Float16*)(ws + alloc(16384*2));
  float*    y2   = (float*)   (ws + alloc(4096*4));
  _Float16* a2h  = (_Float16*)(ws + alloc(4096*2));
  float*    y3   = (float*)   (ws + alloc(128*4));

  // complex GEMM via WMMA: zs[b,o,baseZ + mh*w + nh] = sum_{f,kh} fh * khc
  auto cgemm = [&](const float2* fh, int F, int NSQf, int baseA, int mstride,
                   const float2* khc, int O, int NSQk, int baseB, int nstride,
                   float2* zs, int NSQz, int baseZ, int w, int wk){
    int M  = 128*w;
    int N  = O*w;
    int Nt = (N+15)&~15;
    int K  = 2*F*wk;
    int Kp = (K+31)&~31;
    int TA = M*Kp, TB = Nt*Kp, TC = M*N;
    k_packA<<<dim3((TA+255)/256),dim3(256),0,stream>>>(fh, sAh, F, NSQf, baseA, w, mstride, wk, Kp, TA);
    k_packB<<<dim3((TB+255)/256),dim3(256),0,stream>>>(khc, sBr, F, O, NSQk, baseB, w, nstride, wk, Kp, N, 0, TB);
    k_packB<<<dim3((TB+255)/256),dim3(256),0,stream>>>(khc, sBi, F, O, NSQk, baseB, w, nstride, wk, Kp, N, 1, TB);
    dim3 g(Nt/16, M/64);
    k_gemm_wmma<<<g,dim3(128),0,stream>>>(sAh, sBr, nullptr, sCr, M, N, Kp, 0);
    k_gemm_wmma<<<g,dim3(128),0,stream>>>(sAh, sBi, nullptr, sCi, M, N, Kp, 0);
    k_combine<<<dim3((TC+255)/256),dim3(256),0,stream>>>(sCr, sCi, zs, O, NSQz, baseZ, w, N, TC);
  };

  // ---- constants ----
  {
    float* dsts[10] = {c_w3, c_dl1, (float*)c_bas1, c_dS10, c_dA2,
                       (float*)c_bas2, c_dS6, c_dA3, (float*)c_bas3, c_dS3};
    int    ns[10]   = {6, 2600, 3200, 26600, 5720, 9152, 3432, 420, 1120, 210};
    for (int t=0;t<10;t++)
      k_precompute<<<dim3((ns[t]+127)/128), dim3(128), 0, stream>>>(dsts[t], t, ns[t]);
  }

  // ---- layer 1: s2_conv(b13 -> b10, 34 -> 20ch) ----
  {
    float2* X1 = R2; float2* fh1 = R4a; float2* khc1 = R4b; float2* zs1 = R3; float* h1 = R1;
    int T1 = 128*34*26*26;
    k_fft1<<<dim3((T1+255)/256), dim3(256), 0, stream>>>(x, X1);
    int T2 = 128*34*100;
    k_fh1<<<dim3((T2+255)/256), dim3(256), 0, stream>>>(X1, c_dl1, fh1);
    int T3 = 34*20*100;
    k_khc1<<<dim3((T3+255)/256), dim3(256), 0, stream>>>(k1, c_bas1, khc1, SC1);
    for (int l=0;l<10;l++){
      int w = 2*l+1;
      cgemm(fh1, 34, 100, l*l, /*mstride*/1,
            khc1, 20, 100, l*l, /*nstride*/1,
            zs1, 1330, osq(l), w, /*wk*/1);
    }
    k_synth<20,1330><<<dim3(128*20*20), dim3(128), 0, stream>>>(zs1, c_dS10, b1, h1, 20);
  }

  // ---- layer 2: so3_conv(b10 -> b6, 20 -> 128ch) ----
  {
    float* h1 = R1; float2* Xg2 = R2; float2* fh2 = R4a; float2* khc2 = R4b;
    float2* zs2 = R3; float* h2 = R1;
    k_fft2g<20,11><<<dim3(128*20*20), dim3(128), 0, stream>>>(h1, Xg2);
    int T7 = 128*20*286;
    k_fh23<<<dim3((T7+255)/256), dim3(256), 0, stream>>>(Xg2, c_dA2, fh2, 20, 11, 286, T7);
    int T8 = 20*128*286;
    k_khc23<<<dim3((T8+255)/256), dim3(256), 0, stream>>>(k2, c_bas2, khc2, SC2, 286, T8);
    for (int l=0;l<6;l++){
      int w = 2*l+1;
      cgemm(fh2, 20, 286, osq(l), /*mstride*/w,
            khc2, 128, 286, osq(l), /*nstride*/w,
            zs2, 286, osq(l), w, /*wk*/w);
    }
    k_synth<12,286><<<dim3(128*128*12), dim3(128), 0, stream>>>(zs2, c_dS6, b2, h2, 128);
  }

  // ---- layer 3: so3_conv(b6 -> b3, 128 -> 256ch) ----
  {
    float* h2 = R1; float2* Xg3 = R2; float2* fh3 = R4a; float2* khc3 = R4b;
    float2* zs3 = R3; float* h3 = R1;
    k_fft2g<12,5><<<dim3(128*128*12), dim3(128), 0, stream>>>(h2, Xg3);
    int T12 = 128*128*35;
    k_fh23<<<dim3((T12+255)/256), dim3(256), 0, stream>>>(Xg3, c_dA3, fh3, 12, 5, 35, T12);
    int T13 = 128*256*35;
    k_khc23<<<dim3((T13+255)/256), dim3(256), 0, stream>>>(k3, c_bas3, khc3, SC3, 35, T13);
    for (int l=0;l<3;l++){
      int w = 2*l+1;
      cgemm(fh3, 128, 35, osq(l), /*mstride*/w,
            khc3, 256, 35, osq(l), /*nstride*/w,
            zs3, 35, osq(l), w, /*wk*/w);
    }
    k_synth<6,35><<<dim3(128*256*6), dim3(128), 0, stream>>>(zs3, c_dS3, b3, h3, 256);
  }

  // ---- integrate + MLP ----
  {
    float* h3 = R1;
    int T16 = 128*256;
    k_integrate<<<dim3((T16+255)/256), dim3(256), 0, stream>>>(h3, c_w3, feat);

    k_packW<<<dim3((128*256+255)/256), dim3(256), 0, stream>>>(w1, w1t, 256, 128, 128);
    k_packW<<<dim3((32*128+255)/256),  dim3(256), 0, stream>>>(w2, w2t, 128, 32, 32);
    k_packW<<<dim3((16*32+255)/256),   dim3(256), 0, stream>>>(mw3, w3t, 32, 1, 16);

    // fc1: (128x256)@(256x128) -> bn relu
    k_gemm_wmma<<<dim3(8,2), dim3(128), 0, stream>>>(feat, w1t, bw1, y1, 128, 128, 256, 1);
    k_bn_relu_h<<<dim3(128), dim3(128), 0, stream>>>(y1, g1, be1, a1h, 128);
    // fc2: (128x128)@(128x32) -> bn relu
    k_gemm_wmma<<<dim3(2,2), dim3(128), 0, stream>>>(a1h, w2t, bw2, y2, 128, 32, 128, 1);
    k_bn_relu_h<<<dim3(32), dim3(128), 0, stream>>>(y2, g2, be2, a2h, 32);
    // fc3: (128x32)@(32x1) -> bn sigmoid -> d_out
    k_gemm_wmma<<<dim3(1,2), dim3(128), 0, stream>>>(a2h, w3t, bw3, y3, 128, 1, 32, 1);
    k_bn_sig<<<dim3(1), dim3(128), 0, stream>>>(y3, g3, be3, (float*)d_out, 1);
  }
}
